// MultiHeadAttentionWithRoPE_37890201486118
// MI455X (gfx1250) — matrix operand order repaired and checked by reading, not compiled, not run on hardware
//
#include <hip/hip_runtime.h>

#define S_LEN   4096
#define DMODEL  1024
#define NHEADS  16
#define HDIM    64

typedef __attribute__((ext_vector_type(8)))  float          v8f;
typedef __attribute__((ext_vector_type(16))) __bf16         v16bf;
typedef __attribute__((ext_vector_type(16))) unsigned short v16u;
typedef __attribute__((ext_vector_type(8)))  unsigned short v8u;

union BF16x16 { v16u u; v16bf b; };

__device__ __forceinline__ unsigned short f2bf(float f) {
  unsigned int u = __float_as_uint(f);
  u += 0x7fffu + ((u >> 16) & 1u);          // round-to-nearest-even
  return (unsigned short)(u >> 16);
}

__device__ __forceinline__ v8f wmma_bf16(v16u a, v16u b, v8f c) {
  BF16x16 ua, ub; ua.u = a; ub.u = b;
  // D = A(16x32 bf16) x B(32x16 bf16) + C(16x16 f32)
  return __builtin_amdgcn_wmma_f32_16x16x32_bf16(false, ua.b, false, ub.b,
                                                 (short)0, c, false, false);
}

__device__ __forceinline__ v8f zero8() {
  v8f z;
  #pragma unroll
  for (int i = 0; i < 8; ++i) z[i] = 0.0f;
  return z;
}

// -----------------------------------------------------------------------------
// One-shot fp32 -> bf16 conversion, 8 elements / thread (b128 in, b128 out).
// -----------------------------------------------------------------------------
__global__ void __launch_bounds__(256)
cvt_f32_bf16(const float* __restrict__ src, unsigned short* __restrict__ dst) {
  const size_t i = ((size_t)blockIdx.x * 256 + threadIdx.x) * 8;
  v8f v = *(const v8f*)(src + i);
  v8u o;
  #pragma unroll
  for (int k = 0; k < 8; ++k) o[k] = f2bf(v[k]);
  *(v8u*)(dst + i) = o;
}

// -----------------------------------------------------------------------------
// C[M, N=1024] = A[M, K=1024] (bf16 row-major) x B[N, K] (bf16, torch W) ^T + bias
// One wave computes a 32x64 tile: 2 A-frags x 4 B-frags = 8 WMMAs per K-step.
// All 8 loads of a K-step are issued before the 8-WMMA burst (one clause).
// grid = (N/64, M/32), block = 32 (one wave).
// -----------------------------------------------------------------------------
__global__ void __launch_bounds__(32)
gemm_bf16_xwT_bias(const unsigned short* __restrict__ A,
                   const unsigned short* __restrict__ B,
                   const float* __restrict__ bias, float* __restrict__ C) {
  const int lane  = threadIdx.x;
  const int half  = lane >> 4;        // 0/1
  const int ln16  = lane & 15;
  const int kbase = half * 16;        // A-frag K base within 32-block
  const int khalf = half * 16;        // B-frag K base within 32-block
  const int s0 = blockIdx.y * 32;
  const int e0 = blockIdx.x * 64;

  v8f acc[2][4];
  #pragma unroll
  for (int r = 0; r < 2; ++r)
    #pragma unroll
    for (int c = 0; c < 4; ++c) acc[r][c] = zero8();

  const unsigned short* arow0 = A + (size_t)(s0 + ln16) * DMODEL;
  const unsigned short* arow1 = arow0 + (size_t)16 * DMODEL;

  for (int kb = 0; kb < DMODEL; kb += 32) {
    // A fragments: lane holds row ln16 (+16), K = kbase+e (contiguous)
    v16u a0, a1;
    {
      v8u lo0 = *(const v8u*)(arow0 + kb + kbase);
      v8u hi0 = *(const v8u*)(arow0 + kb + kbase + 8);
      v8u lo1 = *(const v8u*)(arow1 + kb + kbase);
      v8u hi1 = *(const v8u*)(arow1 + kb + kbase + 8);
      #pragma unroll
      for (int i = 0; i < 8; ++i) {
        a0[i] = lo0[i]; a0[i + 8] = hi0[i];
        a1[i] = lo1[i]; a1[i + 8] = hi1[i];
      }
    }
    // B fragments: lane holds col n = ln16, K = khalf + e (contiguous 32B)
    v16u bfr[4];
    #pragma unroll
    for (int c = 0; c < 4; ++c)
      bfr[c] = *(const v16u*)(B + (size_t)(e0 + c * 16 + ln16) * DMODEL + kb + khalf);

    #pragma unroll
    for (int c = 0; c < 4; ++c) {
      acc[0][c] = wmma_bf16(a0, bfr[c], acc[0][c]);
      acc[1][c] = wmma_bf16(a1, bfr[c], acc[1][c]);
    }
  }

  // C/D layout: element v of lane -> (row = half*8+v, col = ln16)
  #pragma unroll
  for (int r = 0; r < 2; ++r) {
    #pragma unroll
    for (int c = 0; c < 4; ++c) {
      const int e = e0 + c * 16 + ln16;
      const float bb = bias[e];
      #pragma unroll
      for (int v = 0; v < 8; ++v)
        C[(size_t)(s0 + r * 16 + half * 8 + v) * DMODEL + e] = acc[r][c][v] + bb;
    }
  }
}

// -----------------------------------------------------------------------------
// RoPE on Q,K + bf16 reshape: Qb/Kb -> (H, S, 64) row-major; V -> Vt (H, 64, S).
// One thread per (s, h, j) with j in [0,32): handles the pair (j, j+32).
// -----------------------------------------------------------------------------
__global__ void __launch_bounds__(256)
rope_reshape(const float* __restrict__ Qf, const float* __restrict__ Kf,
             const float* __restrict__ Vf,
             unsigned short* __restrict__ Qb, unsigned short* __restrict__ Kb,
             unsigned short* __restrict__ Vt) {
  const int idx = blockIdx.x * 256 + threadIdx.x;
  const int j = idx & 31;
  const int h = (idx >> 5) & (NHEADS - 1);
  const int s = idx >> 9;

  const float inv_freq = powf(10000.0f, -(float)j * (1.0f / 32.0f));
  const float ang = (float)s * inv_freq;
  const float cs = cosf(ang);
  const float sn = sinf(ang);

  const size_t src = (size_t)s * DMODEL + h * HDIM + j;
  const float q1 = Qf[src], q2 = Qf[src + 32];
  const float k1 = Kf[src], k2 = Kf[src + 32];

  const size_t dst = ((size_t)h * S_LEN + s) * HDIM + j;
  Qb[dst]      = f2bf(q1 * cs - q2 * sn);
  Qb[dst + 32] = f2bf(q2 * cs + q1 * sn);
  Kb[dst]      = f2bf(k1 * cs - k2 * sn);
  Kb[dst + 32] = f2bf(k2 * cs + k1 * sn);

  const size_t vdst = ((size_t)h * HDIM + j) * S_LEN + s;
  Vt[vdst]                      = f2bf(Vf[src]);
  Vt[vdst + (size_t)32 * S_LEN] = f2bf(Vf[src + 32]);
}

// -----------------------------------------------------------------------------
// Causal flash attention, online softmax. One wave per (head, 32-query block).
// Key blocks of 32: each K-frag / V-frag load feeds TWO WMMAs (row blocks):
// 8 WMMAs for S = Q K^T, 8 WMMAs for O += P V per iteration vs 8 b128 loads.
// P re-layout (C-frag -> A-frag) goes through a 2KB LDS tile.
// Output written directly as bf16 (S, DMODEL) for the final projection GEMM.
// grid = (S/32, H), block = 32.
// -----------------------------------------------------------------------------
__global__ void __launch_bounds__(32)
flash_attn(const unsigned short* __restrict__ Qb,
           const unsigned short* __restrict__ Kb,
           const unsigned short* __restrict__ Vt,
           unsigned short* __restrict__ O) {
  __shared__ __align__(32) unsigned short p_lds[2][16 * 32];

  const int lane  = threadIdx.x;
  const int half  = lane >> 4;
  const int ln16  = lane & 15;
  const int kbase = half * 16;
  const int khalf = half * 16;
  const int h  = blockIdx.y;
  const int q0 = blockIdx.x * 32;

  // Q A-fragments: [row-block r][d-block 0/1]
  v16u aq[2][2];
  #pragma unroll
  for (int r = 0; r < 2; ++r) {
    const unsigned short* qrow =
        Qb + ((size_t)h * S_LEN + q0 + r * 16 + ln16) * HDIM;
    #pragma unroll
    for (int t = 0; t < 2; ++t) {
      v8u lo = *(const v8u*)(qrow + t * 32 + kbase);
      v8u hi = *(const v8u*)(qrow + t * 32 + kbase + 8);
      #pragma unroll
      for (int i = 0; i < 8; ++i) { aq[r][t][i] = lo[i]; aq[r][t][i + 8] = hi[i]; }
    }
  }

  float mrow[2][8], lrow[2][8];
  #pragma unroll
  for (int r = 0; r < 2; ++r)
    #pragma unroll
    for (int v = 0; v < 8; ++v) { mrow[r][v] = -1e30f; lrow[r][v] = 0.0f; }
  v8f acc[2][4];
  #pragma unroll
  for (int r = 0; r < 2; ++r)
    #pragma unroll
    for (int c = 0; c < 4; ++c) acc[r][c] = zero8();

  const float scale = 0.125f;            // 1/sqrt(64)
  const float LOG2E = 1.44269504f;

  for (int kb = 0; kb < q0 + 32; kb += 32) {
    // ---- scores: sc[r][t] = Q-rows(r) x keys tile t (kb+t*16 .. +15) ----
    v8f sc[2][2];
    #pragma unroll
    for (int r = 0; r < 2; ++r)
      #pragma unroll
      for (int t = 0; t < 2; ++t) sc[r][t] = zero8();

    #pragma unroll
    for (int t = 0; t < 2; ++t) {
      const unsigned short* krow =
          Kb + ((size_t)h * S_LEN + kb + t * 16 + ln16) * HDIM;
      #pragma unroll
      for (int db = 0; db < 2; ++db) {
        v16u bk = *(const v16u*)(krow + db * 32 + khalf);
        sc[0][t] = wmma_bf16(aq[0][db], bk, sc[0][t]);
        sc[1][t] = wmma_bf16(aq[1][db], bk, sc[1][t]);
      }
    }

    // ---- mask + online softmax (rows live in fixed 16-lane groups) ----
    #pragma unroll
    for (int r = 0; r < 2; ++r) {
      #pragma unroll
      for (int v = 0; v < 8; ++v) {
        const int qidx = q0 + r * 16 + half * 8 + v;
        float e0 = (kb + ln16      <= qidx) ? sc[r][0][v] * scale : -1e30f;
        float e1 = (kb + 16 + ln16 <= qidx) ? sc[r][1][v] * scale : -1e30f;
        float mx = fmaxf(e0, e1);
        #pragma unroll
        for (int off = 8; off >= 1; off >>= 1) mx = fmaxf(mx, __shfl_xor(mx, off));
        const float mnew  = fmaxf(mrow[r][v], mx);
        const float alpha = exp2f((mrow[r][v] - mnew) * LOG2E);
        mrow[r][v] = mnew;
        const float p0 = exp2f((e0 - mnew) * LOG2E);
        const float p1 = exp2f((e1 - mnew) * LOG2E);
        float rs = p0 + p1;
        #pragma unroll
        for (int off = 8; off >= 1; off >>= 1) rs += __shfl_xor(rs, off);
        lrow[r][v] = lrow[r][v] * alpha + rs;
        #pragma unroll
        for (int c = 0; c < 4; ++c) acc[r][c][v] *= alpha;
        p_lds[r][(half * 8 + v) * 32 + ln16]      = f2bf(p0);
        p_lds[r][(half * 8 + v) * 32 + 16 + ln16] = f2bf(p1);
      }
    }
    __syncthreads();

    // ---- P as A-fragments (16x32 over keys), one per row block ----
    v16u ap[2];
    #pragma unroll
    for (int r = 0; r < 2; ++r) {
      const unsigned short* prow = p_lds[r] + ln16 * 32;
      v8u lo = *(const v8u*)(prow + kbase);
      v8u hi = *(const v8u*)(prow + kbase + 8);
      #pragma unroll
      for (int i = 0; i < 8; ++i) { ap[r][i] = lo[i]; ap[r][i + 8] = hi[i]; }
    }
    __syncthreads();

    // ---- O += P x V : each V B-frag feeds both row blocks ----
    #pragma unroll
    for (int c = 0; c < 4; ++c) {
      const unsigned short* vrow =
          Vt + ((size_t)h * HDIM + c * 16 + ln16) * S_LEN + kb + khalf;
      v16u bv = *(const v16u*)(vrow);
      acc[0][c] = wmma_bf16(ap[0], bv, acc[0][c]);
      acc[1][c] = wmma_bf16(ap[1], bv, acc[1][c]);
    }
  }

  // ---- normalize and store bf16 (S, DMODEL) ----
  #pragma unroll
  for (int r = 0; r < 2; ++r) {
    #pragma unroll
    for (int v = 0; v < 8; ++v) {
      const float inv = 1.0f / lrow[r][v];
      const size_t row =
          (size_t)(q0 + r * 16 + half * 8 + v) * DMODEL + h * HDIM;
      #pragma unroll
      for (int c = 0; c < 4; ++c)
        O[row + c * 16 + ln16] = f2bf(acc[r][c][v] * inv);
    }
  }
}

// -----------------------------------------------------------------------------
extern "C" void kernel_launch(void* const* d_in, const int* in_sizes, int n_in,
                              void* d_out, int out_size, void* d_ws, size_t ws_size,
                              hipStream_t stream) {
  const float* x  = (const float*)d_in[0];
  const float* Wq = (const float*)d_in[1];
  const float* bq = (const float*)d_in[2];
  const float* Wk = (const float*)d_in[3];
  const float* bk = (const float*)d_in[4];
  const float* Wv = (const float*)d_in[5];
  const float* bv = (const float*)d_in[6];
  const float* Wo = (const float*)d_in[7];
  const float* bo = (const float*)d_in[8];
  // d_in[9] = causal mask, implicit in flash_attn
  float* out = (float*)d_out;

  char* ws = (char*)d_ws;
  const size_t SZ_XF = (size_t)S_LEN * DMODEL * sizeof(float);           // 16 MB
  const size_t SZ_XB = (size_t)S_LEN * DMODEL * sizeof(unsigned short);  //  8 MB
  const size_t SZ_WB = (size_t)DMODEL * DMODEL * sizeof(unsigned short); //  2 MB

  unsigned short* xb  = (unsigned short*)(ws);
  unsigned short* Wqb = (unsigned short*)(ws + SZ_XB);
  unsigned short* Wkb = (unsigned short*)(ws + SZ_XB + SZ_WB);
  unsigned short* Wvb = (unsigned short*)(ws + SZ_XB + 2 * SZ_WB);
  unsigned short* Wob = (unsigned short*)(ws + SZ_XB + 3 * SZ_WB);
  char* p = ws + SZ_XB + 4 * SZ_WB;                 // +16 MB
  float* Qf = (float*)(p);
  float* Kf = (float*)(p + SZ_XF);
  float* Vf = (float*)(p + 2 * SZ_XF);
  unsigned short* Qb = (unsigned short*)(p + 3 * SZ_XF);
  unsigned short* Kb = (unsigned short*)(p + 3 * SZ_XF + SZ_XB);
  unsigned short* Vt = (unsigned short*)(p + 3 * SZ_XF + 2 * SZ_XB);
  unsigned short* Ob = (unsigned short*)(Qf);   // reuse: Qf dead after rope_reshape

  // --- one-shot bf16 conversions ---
  const int nx8 = (S_LEN * DMODEL) / 8;   // 524288 threads -> 2048 blocks
  const int nw8 = (DMODEL * DMODEL) / 8;  // 131072 threads ->  512 blocks
  cvt_f32_bf16<<<nx8 / 256, 256, 0, stream>>>(x,  xb);
  cvt_f32_bf16<<<nw8 / 256, 256, 0, stream>>>(Wq, Wqb);
  cvt_f32_bf16<<<nw8 / 256, 256, 0, stream>>>(Wk, Wkb);
  cvt_f32_bf16<<<nw8 / 256, 256, 0, stream>>>(Wv, Wvb);
  cvt_f32_bf16<<<nw8 / 256, 256, 0, stream>>>(Wo, Wob);

  // --- QKV projections ---
  const dim3 gg(DMODEL / 64, S_LEN / 32);   // (16, 128)
  gemm_bf16_xwT_bias<<<gg, 32, 0, stream>>>(xb, Wqb, bq, Qf);
  gemm_bf16_xwT_bias<<<gg, 32, 0, stream>>>(xb, Wkb, bk, Kf);
  gemm_bf16_xwT_bias<<<gg, 32, 0, stream>>>(xb, Wvb, bv, Vf);

  // --- RoPE + head reshape ---
  const int total = S_LEN * NHEADS * 32;    // 2M threads
  rope_reshape<<<total / 256, 256, 0, stream>>>(Qf, Kf, Vf, Qb, Kb, Vt);

  // --- attention ---
  flash_attn<<<dim3(S_LEN / 32, NHEADS), 32, 0, stream>>>(Qb, Kb, Vt, Ob);

  // --- output projection ---
  gemm_bf16_xwT_bias<<<gg, 32, 0, stream>>>(Ob, Wob, bo, out);
}
